// HG_Net_51101520888407
// MI455X (gfx1250) — matrix operand order repaired and checked by reading, not compiled, not run on hardware
//
#include <hip/hip_runtime.h>

#define H   128
#define H3  384
#define NPA 131072
#define NLA 32768
#define EPP 524288
#define ELL 131072
#define ELP 262144
#define EPL 262144
#define BB  512

typedef __attribute__((ext_vector_type(8)))  __bf16 v8bf;
typedef __attribute__((ext_vector_type(16))) __bf16 v16bf;
typedef __attribute__((ext_vector_type(8)))  float  v8f;

// ---------------------------------------------------------------- device utils
__device__ __forceinline__ float leakyf(float x) { return x > 0.f ? x : 0.2f * x; }
__device__ __forceinline__ float sigmf(float x)  { return 1.f / (1.f + __expf(-x)); }

__device__ __forceinline__ void atomicMaxF(float* addr, float val) {
    int* ai = (int*)addr;
    int old = *ai;
    while (true) {
        float f = __int_as_float(old);
        if (f >= val) break;
        int assumed = old;
        old = atomicCAS(ai, assumed, __float_as_int(val));
        if (old == assumed) break;
    }
}

// ---------------------------------------------------------------- WMMA GEMM
// C[M,N] = act( A[M,Kp](bf16,row-major) * Bp(packed bf16) + bias[N] )
// Block = 256 threads = 8 waves; block tile = 128 rows x 128 cols.
// B slice (Kp x 128, <= 32KB) is staged in LDS once and shared by all 8 waves;
// inner loop: ds_load_b128 B fragments + software-pipelined global A loads,
// 8 back-to-back v_wmma_f32_16x16x32_bf16 per K-step.
// M % 128 == 0, N % 128 == 0, Kp % 32 == 0 (guaranteed by caller).
__global__ void __launch_bounds__(256)
k_gemm(const __bf16* __restrict__ A, const __bf16* __restrict__ Bp,
       const float* __restrict__ bias, float* __restrict__ C,
       int M, int Kp, int N, int act)
{
    extern __shared__ __bf16 smem[];          // [Kp/32][8 ntiles][32 lanes][16]
    const int lane   = threadIdx.x & 31;
    const int wave   = threadIdx.x >> 5;
    const int m0     = blockIdx.x * 128 + wave * 16;
    const int n0     = blockIdx.y * 128;
    const int ntiles = N >> 4;
    const int ktiles = Kp >> 5;

    // ---- cooperative stage of the 128-column B slice into LDS
    {
        const int total4 = ktiles * 512;      // 16-byte chunks (8 tiles * 1KB per ktile)
        for (int i = threadIdx.x; i < total4; i += 256) {
            int kt  = i >> 9;
            int rem = i & 511;
            const uint4* sp = (const uint4*)(Bp + ((size_t)kt * ntiles + (n0 >> 4)) * 512) + rem;
            ((uint4*)smem)[i] = *sp;
        }
    }
    __syncthreads();

    const int row = m0 + (lane & 15);
    const int kg  = lane >> 4;
    const __bf16* arow = A + (size_t)row * Kp + kg * 16;

    v8f acc[8] = {};

    // A fragment (16x32 bf16, contiguous k = 16*(lane/16) + s per lane), software-pipelined
    v8bf alo = *(const v8bf*)(arow);
    v8bf ahi = *(const v8bf*)(arow + 8);

    for (int kt = 0; kt < Kp; kt += 32) {
        v16bf af = __builtin_shufflevector(alo, ahi,
                      0,1,2,3,4,5,6,7,8,9,10,11,12,13,14,15);
        if (kt + 32 < Kp) {                  // issue next K-step's A loads early
            alo = *(const v8bf*)(arow + kt + 32);
            ahi = *(const v8bf*)(arow + kt + 40);
        }
        const __bf16* bk = smem + (size_t)(kt >> 5) * 4096 + (size_t)lane * 16;
        v16bf bf[8];
#pragma unroll
        for (int nt = 0; nt < 8; ++nt)
            bf[nt] = *(const v16bf*)(bk + (size_t)nt * 512);
#pragma unroll
        for (int nt = 0; nt < 8; ++nt)
            acc[nt] = __builtin_amdgcn_wmma_f32_16x16x32_bf16(
                          false, af, false, bf[nt], (short)0, acc[nt], false, false);
    }

    // C layout: VGPR r -> row m0 + r (+8 for lanes 16-31); col = n-tile base + lane&15
    const int rbase = m0 + ((lane >> 4) << 3);
#pragma unroll
    for (int nt = 0; nt < 8; ++nt) {
        const int col = n0 + nt * 16 + (lane & 15);
        float bv = bias ? bias[col] : 0.f;
#pragma unroll
        for (int r = 0; r < 8; ++r) {
            float v = acc[nt][r] + bv;
            if (act == 1)      v = fmaxf(v, 0.f);
            else if (act == 2) v = v > 0.f ? v : 0.2f * v;
            C[(size_t)(rbase + r) * N + col] = v;
        }
    }
}

// ---------------------------------------------------------------- pack / convert
// Pack weight [K,N] f32 -> bf16 fragment order [Kp/32][N/16][32 lanes][16 halves]
__global__ void k_pack_w(const float* __restrict__ src, __bf16* __restrict__ dst,
                         int K, int Kp, int N)
{
    int idx = blockIdx.x * 256 + threadIdx.x;
    int total = Kp * N;
    if (idx >= total) return;
    int within = idx & 511;
    int tile   = idx >> 9;
    int nt   = tile % (N >> 4);
    int kt   = tile / (N >> 4);
    int lane = within >> 4;
    int hh   = within & 15;
    int kk   = kt * 32 + ((lane >> 4) << 4) + hh;  // lanes 0-15: K 0..15, 16-31: K 16..31
    int col  = nt * 16 + (lane & 15);
    float v  = (kk < K) ? src[(size_t)kk * N + col] : 0.f;
    dst[idx] = (__bf16)v;
}

__global__ void k_cvt_pad(const float* __restrict__ src, __bf16* __restrict__ dst,
                          long total, int K, int Kp)
{
    long idx = (long)blockIdx.x * 256 + threadIdx.x;
    if (idx >= total) return;
    long r = idx / Kp; int c = (int)(idx % Kp);
    dst[idx] = (c < K) ? (__bf16)src[r * K + c] : (__bf16)0.f;
}

__global__ void k_cvt(const float* __restrict__ src, __bf16* __restrict__ dst, long n)
{
    long i = (long)blockIdx.x * 256 + threadIdx.x;
    if (i < n) dst[i] = (__bf16)src[i];
}

// edge input: [x_src[src[e]] (din) | ea[e] (ed) | zero-pad] -> bf16 [E,96]
__global__ void k_edge_concat(const float* __restrict__ xsrc, const float* __restrict__ ea,
                              const int* __restrict__ srci, __bf16* __restrict__ dst,
                              long total, int din, int ed)
{
    long idx = (long)blockIdx.x * 256 + threadIdx.x;
    if (idx >= total) return;
    long e = idx / 96; int c = (int)(idx % 96);
    float v = 0.f;
    if (c < din)            v = xsrc[(size_t)srci[e] * din + c];
    else if (c < din + ed)  v = ea[(size_t)e * ed + (c - din)];
    dst[idx] = (__bf16)v;
}

// ---------------------------------------------------------------- small kernels
__global__ void k_rowdot(const float* __restrict__ x, const float* __restrict__ w,
                         float* __restrict__ out, int N, int D)
{
    int r = blockIdx.x * 8 + (threadIdx.x >> 5);
    int lane = threadIdx.x & 31;
    if (r >= N) return;
    const float* row = x + (size_t)r * D;
    float s = 0.f;
    for (int c = lane; c < D; c += 32) s += row[c] * w[c];
    for (int off = 16; off > 0; off >>= 1) s += __shfl_xor(s, off, 32);
    if (lane == 0) out[r] = s;
}

__global__ void k_gate_logit(const float* __restrict__ m, const float* __restrict__ att_l,
                             const float* __restrict__ rdst, const int* __restrict__ dsti,
                             float* __restrict__ logit, int E)
{
    int e = blockIdx.x * 8 + (threadIdx.x >> 5);
    int lane = threadIdx.x & 31;
    if (e >= E) return;
    const float* row = m + (size_t)e * H;
    float s = 0.f;
#pragma unroll
    for (int c = 0; c < 4; ++c) s += row[lane + 32 * c] * att_l[lane + 32 * c];
    for (int off = 16; off > 0; off >>= 1) s += __shfl_xor(s, off, 32);
    if (lane == 0) logit[e] = leakyf(s + rdst[dsti[e]]);
}

__global__ void k_gat_logit(const float* __restrict__ rs, const float* __restrict__ rd,
                            const int* __restrict__ srci, const int* __restrict__ dsti,
                            float* __restrict__ logit, int E)
{
    int e = blockIdx.x * 256 + threadIdx.x;
    if (e >= E) return;
    int sv = srci ? srci[e] : e;
    logit[e] = leakyf(rs[sv] + rd[dsti[e]]);
}

__global__ void k_fill(float* p, float v, long n)
{
    long i = (long)blockIdx.x * 256 + threadIdx.x;
    if (i < n) p[i] = v;
}

__global__ void k_segmax(const float* __restrict__ logit, const int* __restrict__ dsti,
                         float* __restrict__ smax, int E)
{
    int e = blockIdx.x * 256 + threadIdx.x;
    if (e < E) atomicMaxF(&smax[dsti[e]], logit[e]);
}

__global__ void k_fixmax(float* smax, int N)
{
    int i = blockIdx.x * 256 + threadIdx.x;
    if (i < N) { float m = smax[i]; if (!__builtin_isfinite(m)) smax[i] = 0.f; }
}

__global__ void k_expsum(float* __restrict__ logit, const float* __restrict__ smax,
                         const int* __restrict__ dsti, float* __restrict__ ssum, int E)
{
    int e = blockIdx.x * 256 + threadIdx.x;
    if (e >= E) return;
    int d = dsti[e];
    float v = __expf(logit[e] - smax[d]);
    logit[e] = v;
    atomicAdd(&ssum[d], v);
}

__global__ void k_norm(float* __restrict__ a, const float* __restrict__ ssum,
                       const int* __restrict__ dsti, int E)
{
    int e = blockIdx.x * 256 + threadIdx.x;
    if (e < E) a[e] = a[e] / (ssum[dsti[e]] + 1e-16f);
}

__global__ void k_init_rows(float* __restrict__ ctx, const float* __restrict__ b, long total)
{
    long i = (long)blockIdx.x * 256 + threadIdx.x;
    if (i < total) ctx[i] = b[i & (H - 1)];
}

// ctx[dst[e]] += a[e] * val[src-or-e]; 4 channels per thread
__global__ void k_scatter(const float* __restrict__ a, const float* __restrict__ val,
                          const int* __restrict__ srci, const int* __restrict__ dsti,
                          float* __restrict__ ctx, long total)
{
    long idx = (long)blockIdx.x * 256 + threadIdx.x;
    if (idx >= total) return;
    long e = idx >> 5; int c4 = ((int)idx & 31) << 2;
    long vr = srci ? (long)srci[e] : e;
    float w = a[e];
    const float* v = val + vr * H + c4;
    float* o = ctx + (size_t)dsti[e] * H + c4;
    atomicAdd(o + 0, w * v[0]);
    atomicAdd(o + 1, w * v[1]);
    atomicAdd(o + 2, w * v[2]);
    atomicAdd(o + 3, w * v[3]);
}

__global__ void k_segpool(const float* __restrict__ h, const int* __restrict__ batch,
                          float* __restrict__ mol, long total)
{
    long idx = (long)blockIdx.x * 256 + threadIdx.x;
    if (idx >= total) return;
    long i = idx >> 5; int c4 = ((int)idx & 31) << 2;
    const float* v = h + i * H + c4;
    float* o = mol + (size_t)batch[i] * H + c4;
    atomicAdd(o + 0, v[0]);
    atomicAdd(o + 1, v[1]);
    atomicAdd(o + 2, v[2]);
    atomicAdd(o + 3, v[3]);
}

__global__ void k_elu_bf(float* __restrict__ ctx, __bf16* __restrict__ xb, long n)
{
    long i = (long)blockIdx.x * 256 + threadIdx.x;
    if (i >= n) return;
    float v = ctx[i];
    float g = v > 0.f ? v : (__expf(v) - 1.f);
    ctx[i] = g;
    xb[i]  = (__bf16)g;
}

__global__ void k_relu_cvt(const float* __restrict__ acc, float* __restrict__ h,
                           __bf16* __restrict__ hb, long n)
{
    long i = (long)blockIdx.x * 256 + threadIdx.x;
    if (i >= n) return;
    float v = fmaxf(acc[i], 0.f);
    h[i]  = v;
    hb[i] = (__bf16)v;
}

__global__ void k_gru(const float* __restrict__ gi, const float* __restrict__ gh,
                      const float* __restrict__ h, float* __restrict__ out,
                      long total, int accumulate)
{
    long idx = (long)blockIdx.x * 256 + threadIdx.x;
    if (idx >= total) return;
    long r = idx >> 7; int c = (int)idx & 127;
    const float* gir = gi + r * H3;
    const float* ghr = gh + r * H3;
    float rr = sigmf(gir[c] + ghr[c]);
    float zz = sigmf(gir[H + c] + ghr[H + c]);
    float nn = tanhf(gir[2 * H + c] + rr * ghr[2 * H + c]);
    float o  = (1.f - zz) * nn + zz * h[idx];
    if (accumulate) out[idx] += o; else out[idx] = o;
}

__global__ void k_head(const float* __restrict__ ypa, const float* __restrict__ yla,
                       const float* __restrict__ W1, const float* __restrict__ b1,
                       const float* __restrict__ W2, const float* __restrict__ b2,
                       float* __restrict__ out, int B_)
{
    int idx = blockIdx.x * 256 + threadIdx.x;
    if (idx >= B_ * 256) return;
    int b = idx >> 8; int j = idx & 255;
    float x = (j < H) ? ypa[b * H + j] : yla[b * H + (j - H)];
    float s = b2[0];
#pragma unroll
    for (int k = 0; k < 32; ++k)
        s += fmaxf(x * W1[k] + b1[k], 0.f) * W2[k];
    out[idx] = s;
}

// ---------------------------------------------------------------- host side
namespace {

inline long cdiv(long a, long b) { return (a + b - 1) / b; }

struct GateW {
    const float *W1, *W2, *Wself, *att_l, *att_r, *b, *Whh, *Wih, *bhh, *bih;
    int din, ed;
    __bf16 *W1p, *W2p, *Wselfp, *Whhp, *Wihp;
};
struct GatW {
    const float *Wdst, *Wsrc, *att_dst, *att_src, *b, *Whh, *Wih, *bhh, *bih;
    __bf16 *Wdstp, *Wsrcp, *Whhp, *Wihp;
};
struct Scratch {
    __bf16* einb; float* ef; __bf16* ebf; float* elog;
    float *rA, *rB, *smax, *ssum;
};

// JAX pytree leaf order = sorted dict keys (recursively).
GateW gate_off(const float* base, size_t& c, int din, int ed) {
    GateW g; g.din = din; g.ed = ed;
    g.W1    = base + c; c += (size_t)(din + ed) * H;
    g.W2    = base + c; c += (size_t)H * H;
    g.Wself = base + c; c += (size_t)din * H;
    g.att_l = base + c; c += H;
    g.att_r = base + c; c += din;
    g.b     = base + c; c += H;
    g.Whh   = base + c; c += (size_t)H * H3;
    g.Wih   = base + c; c += (size_t)H * H3;
    g.bhh   = base + c; c += H3;
    g.bih   = base + c; c += H3;
    return g;
}
GatW gat_off(const float* base, size_t& c) {
    GatW g;
    g.Wdst    = base + c; c += (size_t)H * H;
    g.Wsrc    = base + c; c += (size_t)H * H;
    g.att_dst = base + c; c += H;
    g.att_src = base + c; c += H;
    g.b       = base + c; c += H;
    g.Whh     = base + c; c += (size_t)H * H3;
    g.Wih     = base + c; c += (size_t)H * H3;
    g.bhh     = base + c; c += H3;
    g.bih     = base + c; c += H3;
    return g;
}

struct Alloc {
    char* p; size_t off;
    void* get(size_t bytes) { void* r = p + off; off = (off + bytes + 255) & ~(size_t)255; return r; }
    float*  f(size_t n)  { return (float*)get(n * 4); }
    __bf16* bf(size_t n) { return (__bf16*)get(n * 2); }
};

void gemm(hipStream_t s, const __bf16* A, const __bf16* Bp, const float* bias,
          float* C, int M, int Kp, int N, int act) {
    dim3 g((unsigned)(M / 128), (unsigned)(N / 128));
    size_t shbytes = (size_t)Kp * 256;   // Kp x 128 bf16 slice of B
    k_gemm<<<g, 256, shbytes, s>>>(A, Bp, bias, C, M, Kp, N, act);
}
void packw(hipStream_t s, const float* W, __bf16* d, int K, int Kp, int N) {
    k_pack_w<<<(unsigned)cdiv((long)Kp * N, 256), 256, 0, s>>>(W, d, K, Kp, N);
}
void fillf(hipStream_t s, float* p, float v, long n) {
    k_fill<<<(unsigned)cdiv(n, 256), 256, 0, s>>>(p, v, n);
}

void softmax_seg(hipStream_t s, float* elog, const int* dsti, int E, int Nd, Scratch& S) {
    fillf(s, S.smax, -__builtin_huge_valf(), Nd);
    k_segmax<<<(unsigned)cdiv(E, 256), 256, 0, s>>>(elog, dsti, S.smax, E);
    k_fixmax<<<(unsigned)cdiv(Nd, 256), 256, 0, s>>>(S.smax, Nd);
    fillf(s, S.ssum, 0.f, Nd);
    k_expsum<<<(unsigned)cdiv(E, 256), 256, 0, s>>>(elog, S.smax, dsti, S.ssum, E);
    k_norm<<<(unsigned)cdiv(E, 256), 256, 0, s>>>(elog, S.ssum, dsti, E);
}

void run_gate(hipStream_t s, const GateW& g,
              const float* xsrc, const float* xdst, const __bf16* xdst_bf,
              const float* ea, const int* srci, const int* dsti,
              int E, int Nd,
              float* ctx, __bf16* xgbf, float* h0, __bf16* h0bf,
              float* gi, float* gh, float* acc, int accumulate, Scratch& S)
{
    long nh = (long)Nd * H;
    k_rowdot<<<(unsigned)cdiv(Nd, 8), 256, 0, s>>>(xdst, g.att_r, S.rA, Nd, g.din);
    long tot = (long)E * 96;
    k_edge_concat<<<(unsigned)cdiv(tot, 256), 256, 0, s>>>(xsrc, ea, srci, S.einb, tot, g.din, g.ed);
    gemm(s, S.einb, g.W1p, nullptr, S.ef, E, 96, H, 2);                 // m = leaky(..@W1)
    k_cvt<<<(unsigned)cdiv((long)E * H, 256), 256, 0, s>>>(S.ef, S.ebf, (long)E * H);
    k_gate_logit<<<(unsigned)cdiv(E, 8), 256, 0, s>>>(S.ef, g.att_l, S.rA, dsti, S.elog, E);
    softmax_seg(s, S.elog, dsti, E, Nd, S);
    gemm(s, S.ebf, g.W2p, nullptr, S.ef, E, H, H, 0);                   // mw2 (reuse ef)
    k_init_rows<<<(unsigned)cdiv(nh, 256), 256, 0, s>>>(ctx, g.b, nh);
    k_scatter<<<(unsigned)cdiv((long)E * 32, 256), 256, 0, s>>>(S.elog, S.ef, nullptr, dsti, ctx, (long)E * 32);
    k_elu_bf<<<(unsigned)cdiv(nh, 256), 256, 0, s>>>(ctx, xgbf, nh);
    gemm(s, xdst_bf, g.Wselfp, nullptr, h0, Nd, 64, H, 1);              // h0 = relu(x@Wself)
    k_cvt<<<(unsigned)cdiv(nh, 256), 256, 0, s>>>(h0, h0bf, nh);
    gemm(s, xgbf, g.Wihp, g.bih, gi, Nd, H, H3, 0);
    gemm(s, h0bf, g.Whhp, g.bhh, gh, Nd, H, H3, 0);
    k_gru<<<(unsigned)cdiv(nh, 256), 256, 0, s>>>(gi, gh, h0, acc, nh, accumulate);
}

void run_gat(hipStream_t s, const GatW& g,
             const __bf16* hsrc_bf, int Ns,
             const float* hdst, const __bf16* hdst_bf, int Nd,
             const int* srci, const int* dsti, int E,
             float* kbuf, float* qbuf, float* ctx, __bf16* xgbf,
             float* gi, float* gh, float* acc, int accumulate, Scratch& S)
{
    long nh = (long)Nd * H;
    gemm(s, hsrc_bf, g.Wsrcp, nullptr, kbuf, Ns, H, H, 0);
    gemm(s, hdst_bf, g.Wdstp, nullptr, qbuf, Nd, H, H, 0);
    k_rowdot<<<(unsigned)cdiv(Ns, 8), 256, 0, s>>>(kbuf, g.att_src, S.rA, Ns, H);
    k_rowdot<<<(unsigned)cdiv(Nd, 8), 256, 0, s>>>(qbuf, g.att_dst, S.rB, Nd, H);
    k_gat_logit<<<(unsigned)cdiv(E, 256), 256, 0, s>>>(S.rA, S.rB, srci, dsti, S.elog, E);
    softmax_seg(s, S.elog, dsti, E, Nd, S);
    k_init_rows<<<(unsigned)cdiv(nh, 256), 256, 0, s>>>(ctx, g.b, nh);
    k_scatter<<<(unsigned)cdiv((long)E * 32, 256), 256, 0, s>>>(S.elog, kbuf, srci, dsti, ctx, (long)E * 32);
    k_elu_bf<<<(unsigned)cdiv(nh, 256), 256, 0, s>>>(ctx, xgbf, nh);
    gemm(s, xgbf, g.Wihp, g.bih, gi, Nd, H, H3, 0);
    gemm(s, hdst_bf, g.Whhp, g.bhh, gh, Nd, H, H3, 0);
    k_gru<<<(unsigned)cdiv(nh, 256), 256, 0, s>>>(gi, gh, hdst, acc, nh, accumulate);
}

} // namespace

extern "C" void kernel_launch(void* const* d_in, const int* in_sizes, int n_in,
                              void* d_out, int out_size, void* d_ws, size_t ws_size,
                              hipStream_t stream)
{
    (void)in_sizes; (void)n_in; (void)out_size; (void)ws_size;
    const float* x_pa  = (const float*)d_in[0];
    const float* x_la  = (const float*)d_in[1];
    const float* ea_pp = (const float*)d_in[2];
    const float* ea_ll = (const float*)d_in[3];
    const float* ea_lp = (const float*)d_in[4];
    const float* ea_pl = (const float*)d_in[5];
    const float* Pf    = (const float*)d_in[6];
    const int* ei_pp   = (const int*)d_in[7];
    const int* ei_ll   = (const int*)d_in[8];
    const int* ei_lp   = (const int*)d_in[9];
    const int* ei_pl   = (const int*)d_in[10];
    const int* batch_pa = (const int*)d_in[11];
    const int* batch_la = (const int*)d_in[12];
    float* out = (float*)d_out;
    hipStream_t s = stream;

    // ---- parameter offsets (sorted-key pytree order)
    size_t pc = 0;
    GateW g0_ll = gate_off(Pf, pc, 59, 6);
    GateW g0_lp = gate_off(Pf, pc, 59, 11);
    GateW g0_pl = gate_off(Pf, pc, 59, 11);
    GateW g0_pp = gate_off(Pf, pc, 59, 6);
    GatW g1_ll = gat_off(Pf, pc), g1_lp = gat_off(Pf, pc), g1_pl = gat_off(Pf, pc), g1_pp = gat_off(Pf, pc);
    GatW g2_ll = gat_off(Pf, pc), g2_lp = gat_off(Pf, pc), g2_pl = gat_off(Pf, pc), g2_pp = gat_off(Pf, pc);
    const float* lin_la_W = Pf + pc; pc += (size_t)H * H;
    const float* lin_la_b = Pf + pc; pc += H;
    const float* lin_pa_W = Pf + pc; pc += (size_t)H * H;
    const float* lin_pa_b = Pf + pc; pc += H;
    const float* mlp_W1 = Pf + pc; pc += 32;
    const float* mlp_b1 = Pf + pc; pc += 32;
    const float* mlp_W2 = Pf + pc; pc += 32;
    const float* mlp_b2 = Pf + pc; pc += 1;
    GatW g_mla = gat_off(Pf, pc);
    GatW g_mpa = gat_off(Pf, pc);

    // ---- workspace layout
    Alloc A{(char*)d_ws, 0};
    __bf16* xpa_bf = A.bf((size_t)NPA * 64);
    __bf16* xla_bf = A.bf((size_t)NLA * 64);
    float *hpa = A.f((size_t)NPA * H), *hla = A.f((size_t)NLA * H);
    __bf16 *hpa_bf = A.bf((size_t)NPA * H), *hla_bf = A.bf((size_t)NLA * H);
    float *accpa = A.f((size_t)NPA * H), *accla = A.f((size_t)NLA * H);
    float *ctxpa = A.f((size_t)NPA * H), *ctxla = A.f((size_t)NLA * H);
    __bf16 *xgpa_bf = A.bf((size_t)NPA * H), *xgla_bf = A.bf((size_t)NLA * H);
    float *s1pa = A.f((size_t)NPA * H), *s2pa = A.f((size_t)NPA * H);
    float *s1la = A.f((size_t)NLA * H), *s2la = A.f((size_t)NLA * H);
    __bf16 *s1bfpa = A.bf((size_t)NPA * H), *s1bfla = A.bf((size_t)NLA * H);
    float *gipa = A.f((size_t)NPA * H3), *ghpa = A.f((size_t)NPA * H3);
    float *gila = A.f((size_t)NLA * H3), *ghla = A.f((size_t)NLA * H3);
    float *molpa = A.f((size_t)BB * H), *molla = A.f((size_t)BB * H);
    __bf16 *molpabf = A.bf((size_t)BB * H), *mollabf = A.bf((size_t)BB * H);
    float *qmol = A.f((size_t)BB * H), *ctxmol = A.f((size_t)BB * H);
    __bf16 *xgmolbf = A.bf((size_t)BB * H);
    float *gimol = A.f((size_t)BB * H3), *ghmol = A.f((size_t)BB * H3);
    float *ypa = A.f((size_t)BB * H), *yla = A.f((size_t)BB * H);
    Scratch S;
    S.einb = A.bf((size_t)EPP * 96);
    S.ef   = A.f((size_t)EPP * H);
    S.ebf  = A.bf((size_t)EPP * H);
    S.elog = A.f(EPP);
    S.rA = A.f(NPA); S.rB = A.f(NPA);
    S.smax = A.f(NPA); S.ssum = A.f(NPA);

    // ---- pack weights (bf16, WMMA B-fragment order)
    auto prep_gate = [&](GateW& g) {
        g.W1p    = A.bf((size_t)96 * H);  packw(s, g.W1,    g.W1p,    g.din + g.ed, 96, H);
        g.W2p    = A.bf((size_t)H * H);   packw(s, g.W2,    g.W2p,    H, H, H);
        g.Wselfp = A.bf((size_t)64 * H);  packw(s, g.Wself, g.Wselfp, g.din, 64, H);
        g.Whhp   = A.bf((size_t)H * H3);  packw(s, g.Whh,   g.Whhp,   H, H, H3);
        g.Wihp   = A.bf((size_t)H * H3);  packw(s, g.Wih,   g.Wihp,   H, H, H3);
    };
    auto prep_gat = [&](GatW& g) {
        g.Wdstp = A.bf((size_t)H * H);  packw(s, g.Wdst, g.Wdstp, H, H, H);
        g.Wsrcp = A.bf((size_t)H * H);  packw(s, g.Wsrc, g.Wsrcp, H, H, H);
        g.Whhp  = A.bf((size_t)H * H3); packw(s, g.Whh,  g.Whhp,  H, H, H3);
        g.Wihp  = A.bf((size_t)H * H3); packw(s, g.Wih,  g.Wihp,  H, H, H3);
    };
    prep_gate(g0_ll); prep_gate(g0_lp); prep_gate(g0_pl); prep_gate(g0_pp);
    prep_gat(g1_ll); prep_gat(g1_lp); prep_gat(g1_pl); prep_gat(g1_pp);
    prep_gat(g2_ll); prep_gat(g2_lp); prep_gat(g2_pl); prep_gat(g2_pp);
    prep_gat(g_mla); prep_gat(g_mpa);
    __bf16* linpaWp = A.bf((size_t)H * H); packw(s, lin_pa_W, linpaWp, H, H, H);
    __bf16* linlaWp = A.bf((size_t)H * H); packw(s, lin_la_W, linlaWp, H, H, H);

    // ---- input features -> padded bf16 (59 -> 64)
    k_cvt_pad<<<(unsigned)cdiv((long)NPA * 64, 256), 256, 0, s>>>(x_pa, xpa_bf, (long)NPA * 64, 59, 64);
    k_cvt_pad<<<(unsigned)cdiv((long)NLA * 64, 256), 256, 0, s>>>(x_la, xla_bf, (long)NLA * 64, 59, 64);

    // ---- layer 0: gated edge-attr convs, hetero sum, relu
    run_gate(s, g0_pp, x_pa, x_pa, xpa_bf, ea_pp, ei_pp, ei_pp + EPP, EPP, NPA,
             ctxpa, xgpa_bf, s1pa, s1bfpa, gipa, ghpa, accpa, 0, S);
    run_gate(s, g0_lp, x_la, x_pa, xpa_bf, ea_lp, ei_lp, ei_lp + ELP, ELP, NPA,
             ctxpa, xgpa_bf, s1pa, s1bfpa, gipa, ghpa, accpa, 1, S);
    run_gate(s, g0_ll, x_la, x_la, xla_bf, ea_ll, ei_ll, ei_ll + ELL, ELL, NLA,
             ctxla, xgla_bf, s1la, s1bfla, gila, ghla, accla, 0, S);
    run_gate(s, g0_pl, x_pa, x_la, xla_bf, ea_pl, ei_pl, ei_pl + EPL, EPL, NLA,
             ctxla, xgla_bf, s1la, s1bfla, gila, ghla, accla, 1, S);
    k_relu_cvt<<<(unsigned)cdiv((long)NPA * H, 256), 256, 0, s>>>(accpa, hpa, hpa_bf, (long)NPA * H);
    k_relu_cvt<<<(unsigned)cdiv((long)NLA * H, 256), 256, 0, s>>>(accla, hla, hla_bf, (long)NLA * H);

    // ---- layers 1..2: GAT + GRU, hetero sum, relu
    const GatW* L[2][4] = { { &g1_pp, &g1_lp, &g1_ll, &g1_pl },
                            { &g2_pp, &g2_lp, &g2_ll, &g2_pl } };
    for (int li = 0; li < 2; ++li) {
        run_gat(s, *L[li][0], hpa_bf, NPA, hpa, hpa_bf, NPA, ei_pp, ei_pp + EPP, EPP,
                s1pa, s2pa, ctxpa, xgpa_bf, gipa, ghpa, accpa, 0, S);
        run_gat(s, *L[li][1], hla_bf, NLA, hpa, hpa_bf, NPA, ei_lp, ei_lp + ELP, ELP,
                s1la, s2pa, ctxpa, xgpa_bf, gipa, ghpa, accpa, 1, S);
        run_gat(s, *L[li][2], hla_bf, NLA, hla, hla_bf, NLA, ei_ll, ei_ll + ELL, ELL,
                s1la, s2la, ctxla, xgla_bf, gila, ghla, accla, 0, S);
        run_gat(s, *L[li][3], hpa_bf, NPA, hla, hla_bf, NLA, ei_pl, ei_pl + EPL, EPL,
                s1pa, s2la, ctxla, xgla_bf, gila, ghla, accla, 1, S);
        k_relu_cvt<<<(unsigned)cdiv((long)NPA * H, 256), 256, 0, s>>>(accpa, hpa, hpa_bf, (long)NPA * H);
        k_relu_cvt<<<(unsigned)cdiv((long)NLA * H, 256), 256, 0, s>>>(accla, hla, hla_bf, (long)NLA * H);
    }

    // ---- molecular pooling + 3 super-node GAT iterations
    fillf(s, molpa, 0.f, (long)BB * H);
    k_segpool<<<(unsigned)cdiv((long)NPA * 32, 256), 256, 0, s>>>(hpa, batch_pa, molpa, (long)NPA * 32);
    fillf(s, molla, 0.f, (long)BB * H);
    k_segpool<<<(unsigned)cdiv((long)NLA * 32, 256), 256, 0, s>>>(hla, batch_la, molla, (long)NLA * 32);
    for (int t = 0; t < 3; ++t) {
        k_cvt<<<(unsigned)cdiv((long)BB * H, 256), 256, 0, s>>>(molpa, molpabf, (long)BB * H);
        run_gat(s, g_mpa, hpa_bf, NPA, molpa, molpabf, BB, nullptr, batch_pa, NPA,
                s1pa, qmol, ctxmol, xgmolbf, gimol, ghmol, molpa, 0, S);
        k_cvt<<<(unsigned)cdiv((long)BB * H, 256), 256, 0, s>>>(molla, mollabf, (long)BB * H);
        run_gat(s, g_mla, hla_bf, NLA, molla, mollabf, BB, nullptr, batch_la, NLA,
                s1la, qmol, ctxmol, xgmolbf, gimol, ghmol, molla, 0, S);
    }

    // ---- heads
    k_cvt<<<(unsigned)cdiv((long)BB * H, 256), 256, 0, s>>>(molpa, molpabf, (long)BB * H);
    k_cvt<<<(unsigned)cdiv((long)BB * H, 256), 256, 0, s>>>(molla, mollabf, (long)BB * H);
    gemm(s, molpabf, linpaWp, lin_pa_b, ypa, BB, H, H, 0);
    gemm(s, mollabf, linlaWp, lin_la_b, yla, BB, H, H, 0);
    k_head<<<(unsigned)cdiv((long)BB * 256, 256), 256, 0, s>>>(ypa, yla, mlp_W1, mlp_b1, mlp_W2, mlp_b2, out, BB);
}
